// AttentionHead_25116968747584
// MI455X (gfx1250) — matrix operand order repaired and checked by reading, not compiled, not run on hardware
//
#include <hip/hip_runtime.h>
#include <hip/hip_bf16.h>

// ---------------- problem constants ----------------
#define BB 32
#define SS 1024
#define DD 1024
#define FF 64
#define BSROWS (BB * SS)   // 32768 flattened rows
#define MT 128             // rows per workgroup (kernel 1)
#define KC 32              // K-chunk (inner dim per WMMA pass)

typedef __attribute__((ext_vector_type(16))) _Float16 v16h;
typedef __attribute__((ext_vector_type(8)))  _Float16 v8h;
typedef __attribute__((ext_vector_type(8)))  float    v8f;

union V16U { v16h v; v8h h[2]; };

// Build a 16-element f16 fragment from two contiguous 8-element (16B) pieces.
__device__ __forceinline__ v16h ldfrag(const _Float16* p0, const _Float16* p1) {
    V16U u;
    u.h[0] = *(const v8h*)p0;
    u.h[1] = *(const v8h*)p1;
    return u.v;
}

__device__ __forceinline__ v8f wmma16(v16h a, v16h b, v8f c) {
    // D(f32 16x16) = A(f16 16x32) * B(f16 32x16) + C
    return __builtin_amdgcn_wmma_f32_16x16x32_f16(
        /*neg_a=*/false, a, /*neg_b=*/false, b,
        /*c_mod=*/(short)0, c, /*reuse_a=*/false, /*reuse_b=*/false);
}

// =====================================================================
// Kernel 1: fused QKV projection.  x:[BS,D] f32 -> q,k,v:[BS,F] f16
// q is pre-scaled by 1/sqrt(D) = 1/32.
// =====================================================================
__global__ void __launch_bounds__(256)
qkv_proj_kernel(const float* __restrict__ x,
                const float* __restrict__ Wq, const float* __restrict__ bq,
                const float* __restrict__ Wk, const float* __restrict__ bk,
                const float* __restrict__ Wv, const float* __restrict__ bv,
                _Float16* __restrict__ qo, _Float16* __restrict__ ko,
                _Float16* __restrict__ vo)
{
    __shared__ _Float16 xt[MT][KC];        // x tile   [row][k]       8 KB
    __shared__ _Float16 wt[3][FF][KC];     // W^T tile [mat][f][k]   12 KB

    const int t    = threadIdx.x;
    const int lane = t & 31;
    const int wid  = t >> 5;               // wave 0..7 -> 16 rows each
    const int hs   = lane >> 4;            // half: lanes 0-15 -> 0, 16-31 -> 1
    const int ln   = lane & 15;
    const int m0   = blockIdx.x * MT;

    v8f acc[12];                           // [mat(3)][ntile(4)] 16x16 f32
    #pragma unroll
    for (int i = 0; i < 12; ++i)
        #pragma unroll
        for (int j = 0; j < 8; ++j) acc[i][j] = 0.0f;

    const float* Ws[3] = {Wq, Wk, Wv};

    for (int kc = 0; kc < DD / KC; ++kc) {
        // ---- stage x tile (f32 -> f16), coalesced ----
        #pragma unroll
        for (int it = 0; it < (MT * KC) / 256; ++it) {
            int idx = it * 256 + t;
            int r = idx >> 5, c = idx & 31;
            xt[r][c] = (_Float16)x[(size_t)(m0 + r) * DD + kc * KC + c];
        }
        // ---- stage W chunks transposed: wt[mat][f][k] ----
        #pragma unroll
        for (int w3 = 0; w3 < 3; ++w3) {
            #pragma unroll
            for (int it = 0; it < (KC * FF) / 256; ++it) {
                int idx = it * 256 + t;
                int d = idx >> 6, f = idx & 63;
                wt[w3][f][d] = (_Float16)Ws[w3][(size_t)(kc * KC + d) * FF + f];
            }
        }
        __syncthreads();

        // A fragment: row m = ln, k contiguous (16 elems)
        const _Float16* ap = &xt[wid * 16 + ln][hs * 16];
        v16h afrag = ldfrag(ap, ap + 8);

        #pragma unroll
        for (int w3 = 0; w3 < 3; ++w3) {
            #pragma unroll
            for (int nt = 0; nt < 4; ++nt) {
                // B fragment: col n = nt*16+ln, k contiguous (16 elems)
                const _Float16* bp = &wt[w3][nt * 16 + ln][hs * 16];
                v16h bfrag = ldfrag(bp, bp + 8);
                acc[w3 * 4 + nt] = wmma16(afrag, bfrag, acc[w3 * 4 + nt]);
            }
        }
        __syncthreads();
    }

    // ---- epilogue: bias add, scale q by 1/32, store f16 ----
    const float*  bs[3]   = {bq, bk, bv};
    _Float16*     outs[3] = {qo, ko, vo};
    #pragma unroll
    for (int w3 = 0; w3 < 3; ++w3) {
        const float scale = (w3 == 0) ? 0.03125f : 1.0f;
        #pragma unroll
        for (int nt = 0; nt < 4; ++nt) {
            int f = nt * 16 + ln;
            float bias = bs[w3][f];
            #pragma unroll
            for (int r = 0; r < 8; ++r) {
                int m = m0 + wid * 16 + r + hs * 8;   // C layout: VGPR r, half
                float vv = (acc[w3 * 4 + nt][r] + bias) * scale;
                outs[w3][(size_t)m * FF + f] = (_Float16)vv;
            }
        }
    }
}

// =====================================================================
// Kernel 2: flash attention.  q,k,v:[BS,F] f16 -> out:[BS,F] f32
// grid = (S/128, B), 8 waves x 16 query rows, 32-key blocks, online softmax.
// =====================================================================
__global__ void __launch_bounds__(256)
flash_attn_kernel(const _Float16* __restrict__ qm,
                  const _Float16* __restrict__ km,
                  const _Float16* __restrict__ vm,
                  float* __restrict__ out)
{
    __shared__ _Float16 Kt[32][FF];        // [key][f]          4 KB
    __shared__ _Float16 Vt[FF][32];        // [f][key] (transp) 4 KB
    __shared__ _Float16 Pst[8][16][32];    // per-wave P stage  8 KB

    const int t    = threadIdx.x;
    const int lane = t & 31;
    const int wid  = t >> 5;
    const int hs   = lane >> 4;
    const int ln   = lane & 15;

    const int    b       = blockIdx.y;
    const int    q0      = blockIdx.x * 128;
    const int    qbase   = q0 + wid * 16;            // wave's first query row
    const size_t rowbase = (size_t)b * SS;

    // ---- load this wave's Q fragments (f16, pre-scaled by 1/32) ----
    v16h qf[2];
    {
        const _Float16* qrow = qm + (rowbase + qbase + ln) * FF;
        #pragma unroll
        for (int c = 0; c < 2; ++c) {
            const _Float16* p = qrow + c * 32 + hs * 16;
            qf[c] = ldfrag(p, p + 8);
        }
    }

    v8f o[4];
    #pragma unroll
    for (int i = 0; i < 4; ++i)
        #pragma unroll
        for (int j = 0; j < 8; ++j) o[i][j] = 0.0f;

    float mrow[8], lrow[8];
    #pragma unroll
    for (int r = 0; r < 8; ++r) { mrow[r] = -__builtin_inff(); lrow[r] = 0.0f; }

    const int nkb = blockIdx.x * 4 + 4;    // causal: key blocks 0..nkb-1

    for (int kb = 0; kb < nkb; ++kb) {
        // ---- cooperative K/V block load (V transposed in LDS) ----
        #pragma unroll
        for (int it = 0; it < (32 * FF) / 256; ++it) {
            int idx = it * 256 + t;
            int key = idx >> 6, f = idx & 63;
            size_t g = (rowbase + kb * 32 + key) * FF + f;
            Kt[key][f] = km[g];
            Vt[f][key] = vm[g];
        }
        __syncthreads();

        const bool active = (kb * 32 <= qbase + 15);
        if (active) {
            // ---- scores: 16 q-rows x 32 keys, inner = F in 2 chunks ----
            v8f s0, s1;
            #pragma unroll
            for (int j = 0; j < 8; ++j) { s0[j] = 0.0f; s1[j] = 0.0f; }
            #pragma unroll
            for (int c = 0; c < 2; ++c) {
                const _Float16* bp0 = &Kt[ln][c * 32 + hs * 16];
                const _Float16* bp1 = &Kt[16 + ln][c * 32 + hs * 16];
                v16h b0 = ldfrag(bp0, bp0 + 8);
                v16h b1 = ldfrag(bp1, bp1 + 8);
                s0 = wmma16(qf[c], b0, s0);
                s1 = wmma16(qf[c], b1, s1);
            }
            // ---- causal mask + online softmax (per C-layout row) ----
            #pragma unroll
            for (int r = 0; r < 8; ++r) {
                int mg = qbase + r + hs * 8;          // global query row
                int k0 = kb * 32 + ln;                // key of s0 lane col
                int k1 = k0 + 16;                     // key of s1 lane col
                float v0 = (k0 <= mg) ? s0[r] : -__builtin_inff();
                float v1 = (k1 <= mg) ? s1[r] : -__builtin_inff();
                float rm = fmaxf(v0, v1);
                rm = fmaxf(rm, __shfl_xor(rm, 1, 16));
                rm = fmaxf(rm, __shfl_xor(rm, 2, 16));
                rm = fmaxf(rm, __shfl_xor(rm, 4, 16));
                rm = fmaxf(rm, __shfl_xor(rm, 8, 16));
                float mnew  = fmaxf(mrow[r], rm);
                float alpha = __expf(mrow[r] - mnew);
                float p0 = __expf(v0 - mnew);
                float p1 = __expf(v1 - mnew);
                float rs = p0 + p1;
                rs += __shfl_xor(rs, 1, 16);
                rs += __shfl_xor(rs, 2, 16);
                rs += __shfl_xor(rs, 4, 16);
                rs += __shfl_xor(rs, 8, 16);
                lrow[r] = lrow[r] * alpha + rs;
                mrow[r] = mnew;
                #pragma unroll
                for (int nt = 0; nt < 4; ++nt) o[nt][r] *= alpha;
                // stage P into A-layout staging buffer
                int prow = r + hs * 8;
                Pst[wid][prow][ln]      = (_Float16)p0;
                Pst[wid][prow][16 + ln] = (_Float16)p1;
            }
        }
        __syncthreads();   // P staging visible to whole wave (uniform barrier)

        if (active) {
            // ---- O += P(16x32) x V(32x64) ----
            const _Float16* pp = &Pst[wid][ln][hs * 16];
            v16h pf = ldfrag(pp, pp + 8);
            #pragma unroll
            for (int nt = 0; nt < 4; ++nt) {
                const _Float16* vp = &Vt[nt * 16 + ln][hs * 16];
                v16h bvf = ldfrag(vp, vp + 8);
                o[nt] = wmma16(pf, bvf, o[nt]);
            }
        }
        __syncthreads();   // before next block overwrites Kt/Vt
    }

    // ---- finalize: divide by softmax denom, store f32 ----
    #pragma unroll
    for (int nt = 0; nt < 4; ++nt) {
        int f = nt * 16 + ln;
        #pragma unroll
        for (int r = 0; r < 8; ++r) {
            int mg = qbase + r + hs * 8;
            out[(rowbase + mg) * FF + f] = o[nt][r] / lrow[r];
        }
    }
}

// =====================================================================
extern "C" void kernel_launch(void* const* d_in, const int* in_sizes, int n_in,
                              void* d_out, int out_size, void* d_ws, size_t ws_size,
                              hipStream_t stream)
{
    (void)in_sizes; (void)n_in; (void)out_size; (void)ws_size;

    const float* x  = (const float*)d_in[0];
    const float* Wq = (const float*)d_in[1];
    const float* bq = (const float*)d_in[2];
    const float* Wk = (const float*)d_in[3];
    const float* bk = (const float*)d_in[4];
    const float* Wv = (const float*)d_in[5];
    const float* bv = (const float*)d_in[6];
    float* out = (float*)d_out;

    // workspace: f16 Q,K,V  (3 * 32768 * 64 * 2B = 12 MB)
    _Float16* qw = (_Float16*)d_ws;
    _Float16* kw = qw + (size_t)BSROWS * FF;
    _Float16* vw = kw + (size_t)BSROWS * FF;

    qkv_proj_kernel<<<dim3(BSROWS / MT), 256, 0, stream>>>(
        x, Wq, bq, Wk, bk, Wv, bv, qw, kw, vw);

    flash_attn_kernel<<<dim3(SS / 128, BB), 256, 0, stream>>>(
        qw, kw, vw, out);
}
